// CFRCausalRAG_78520592105868
// MI455X (gfx1250) — compile-verified, matrix operand order checked by reading, not run on hardware
//
#include <hip/hip_runtime.h>

typedef __attribute__((ext_vector_type(16))) __bf16 v16bf;
typedef __attribute__((ext_vector_type(8)))  float  v8f;

#define NEG_INF (-3.0e38f)
#define NSPLIT 100              // 200000 = 100 * 2000 corpus rows per split
#define ROWS_PER_SPLIT 2000
#define TILES_PER_SPLIT 125     // 2000 / 16

__device__ __forceinline__ unsigned short f2bf(float f) {
  union { float f; unsigned u; } v; v.f = f;
  unsigned u = v.u;
  u += 0x7fffu + ((u >> 16) & 1u);   // round-to-nearest-even
  return (unsigned short)(u >> 16);
}

// ---------------- generic f32 -> bf16 ----------------
__global__ void k_f2bf(const float* __restrict__ src, unsigned short* __restrict__ dst, int n) {
  int i = blockIdx.x * blockDim.x + threadIdx.x;
  if (i < n) dst[i] = f2bf(src[i]);
}

// ---------------- corpus: bf16 copy + 1/max(||row||,eps) ----------------
__global__ __launch_bounds__(64) void k_corpus_prep(const float* __restrict__ corpus,
                                                    unsigned short* __restrict__ cbf,
                                                    float* __restrict__ invn) {
  const int row = blockIdx.x, t = threadIdx.x;
  const float4 v = *(const float4*)(corpus + (size_t)row * 256 + t * 4);
  __shared__ float red[64];
  red[t] = v.x * v.x + v.y * v.y + v.z * v.z + v.w * v.w;
  __syncthreads();
  for (int off = 32; off > 0; off >>= 1) { if (t < off) red[t] += red[t + off]; __syncthreads(); }
  if (t == 0) invn[row] = 1.0f / fmaxf(sqrtf(red[0]), 1e-12f);
  unsigned short* d = cbf + (size_t)row * 256 + t * 4;
  d[0] = f2bf(v.x); d[1] = f2bf(v.y); d[2] = f2bf(v.z); d[3] = f2bf(v.w);
}

// ---------------- patient embedding: linear + l2norm -> bf16 ----------------
__global__ __launch_bounds__(256) void k_patient_emb(const float* __restrict__ patient,
                                                     const float* __restrict__ pe_w,
                                                     const float* __restrict__ pe_b,
                                                     unsigned short* __restrict__ pebf) {
  const int b = blockIdx.x, o = threadIdx.x;
  __shared__ float sp[266];
  __shared__ float red[256];
  for (int i = o; i < 266; i += 256) sp[i] = patient[(size_t)b * 266 + i];
  __syncthreads();
  const float* w = pe_w + (size_t)o * 266;
  float s = pe_b[o];
  for (int k = 0; k < 266; k++) s += w[k] * sp[k];
  red[o] = s * s; __syncthreads();
  for (int off = 128; off > 0; off >>= 1) { if (o < off) red[o] += red[o + off]; __syncthreads(); }
  const float inv = 1.0f / fmaxf(sqrtf(red[0]), 1e-12f);
  pebf[(size_t)b * 256 + o] = f2bf(s * inv);
}

// ---------------- fused similarity GEMM (bf16 WMMA) + per-split top-16 ----------------
// grid = 100 corpus-splits x 16 patient-groups; block = 4 waves, each wave owns a 16-patient tile.
// Per tile of 16 corpus rows: 8x v_wmma_f32_16x16x32_bf16 over K=256 as TWO independent
// accumulation chains (ILP), scale by inv-norm, prefiltered per-lane running top-16.
__global__ __launch_bounds__(128) void k_sim_topk(const unsigned short* __restrict__ pebf,
                                                  const unsigned short* __restrict__ cbf,
                                                  const float* __restrict__ invn,
                                                  float* __restrict__ pscore,
                                                  int* __restrict__ pidx) {
  const int lane = threadIdx.x & 31;
  const int wv   = threadIdx.x >> 5;
  const int split = blockIdx.x >> 4;   // 0..99
  const int ptg   = blockIdx.x & 15;
  const int pt    = ptg * 4 + wv;      // 0..63 patient tiles
  const int m0    = pt * 16;
  const int mrow  = lane & 15;
  const int koff  = (lane < 16) ? 0 : 16;
  const int hi    = (lane < 16) ? 0 : 8;
  const int cb    = (lane < 16) ? 0 : 8;

  // A: 16 patients x 256 (bf16); same K-permutation applied to A and B -> dot invariant
  v16bf a[8];
  {
    const unsigned short* ar = pebf + (size_t)(m0 + mrow) * 256 + koff;
#pragma unroll
    for (int c = 0; c < 8; c++) a[c] = *(const v16bf*)(ar + c * 32);
  }

  float ts[16]; int ti[16];
#pragma unroll
  for (int q = 0; q < 16; q++) { ts[q] = NEG_INF; ti[q] = 0; }
  float tmin = NEG_INF; int tminpos = 0;

  __shared__ float sc[4][16][16];
  const int nbase = split * ROWS_PER_SPLIT;

  for (int t = 0; t < TILES_PER_SPLIT; t++) {
    const int n0 = nbase + t * 16;
    const unsigned short* br = cbf + (size_t)(n0 + mrow) * 256 + koff;
    v8f acc0 = {}, acc1 = {};
#pragma unroll
    for (int c = 0; c < 4; c++) {
      v16bf bm0 = *(const v16bf*)(br + c * 32);
      v16bf bm1 = *(const v16bf*)(br + (c + 4) * 32);
      acc0 = __builtin_amdgcn_wmma_f32_16x16x32_bf16(false, a[c], false, bm0,
                                                     (short)0, acc0, false, false);
      acc1 = __builtin_amdgcn_wmma_f32_16x16x32_bf16(false, a[c + 4], false, bm1,
                                                     (short)0, acc1, false, false);
    }
    const float s = invn[n0 + mrow];
#pragma unroll
    for (int r = 0; r < 8; r++) sc[wv][r + hi][mrow] = (acc0[r] + acc1[r]) * s;  // D: lane=N, vgpr=M
    // same-wave LDS is in-order; lane scans the 8 columns of its patient row
    float vj[8];
#pragma unroll
    for (int j = 0; j < 8; j++) vj[j] = sc[wv][mrow][cb + j];
    float mx = vj[0];
#pragma unroll
    for (int j = 1; j < 8; j++) mx = fmaxf(mx, vj[j]);
    if (mx > tmin) {                 // rare after warm-up: single branch prefilter
#pragma unroll
      for (int j = 0; j < 8; j++) {
        const float v = vj[j];
        if (v > tmin) {
          ts[tminpos] = v; ti[tminpos] = n0 + cb + j;
          tmin = ts[0]; tminpos = 0;
#pragma unroll
          for (int q = 1; q < 16; q++) if (ts[q] < tmin) { tmin = ts[q]; tminpos = q; }
        }
      }
    }
  }

  __shared__ float ms[4][32][16];
  __shared__ int   mi[4][32][16];
#pragma unroll
  for (int q = 0; q < 16; q++) { ms[wv][lane][q] = ts[q]; mi[wv][lane][q] = ti[q]; }
  if (lane < 16) {   // merge the two half-row lanes -> sorted-descending partial top-16
    const int p = m0 + lane;
    float* os = pscore + ((size_t)p * NSPLIT + split) * 16;
    int*   oi = pidx   + ((size_t)p * NSPLIT + split) * 16;
    for (int r = 0; r < 16; r++) {
      float best = NEG_INF; int bl = lane, bq = 0;
      for (int l2 = 0; l2 < 2; l2++) {
        const int ln = lane + l2 * 16;
        for (int q = 0; q < 16; q++) {
          const float v = ms[wv][ln][q];
          if (v > best) { best = v; bl = ln; bq = q; }
        }
      }
      os[r] = best; oi[r] = mi[wv][bl][bq];
      ms[wv][bl][bq] = NEG_INF;
    }
  }
}

// ---------------- merge NSPLIT partial top-16s -> final sorted top-16 ----------------
__global__ __launch_bounds__(256) void k_topk_merge(const float* __restrict__ pscore,
                                                    const int* __restrict__ pidx,
                                                    float* __restrict__ out_scores,
                                                    int* __restrict__ out_idx) {
  const int b = blockIdx.x, t = threadIdx.x;
  const int NC = NSPLIT * 16;
  __shared__ float s[NSPLIT * 16]; __shared__ int si[NSPLIT * 16];
  __shared__ float rs[256]; __shared__ int ri[256];
  for (int i = t; i < NC; i += 256) { s[i] = pscore[(size_t)b * NC + i]; si[i] = pidx[(size_t)b * NC + i]; }
  __syncthreads();
  for (int r = 0; r < 16; r++) {
    float best = NEG_INF; int bp = -1;
    for (int i = t; i < NC; i += 256) if (s[i] > best) { best = s[i]; bp = i; }
    rs[t] = best; ri[t] = bp;
    __syncthreads();
    for (int off = 128; off > 0; off >>= 1) {
      if (t < off && rs[t + off] > rs[t]) { rs[t] = rs[t + off]; ri[t] = ri[t + off]; }
      __syncthreads();
    }
    if (t == 0) {
      const int p = ri[0];
      out_scores[(size_t)b * 16 + r] = rs[0];
      out_idx[(size_t)b * 16 + r] = (p >= 0) ? si[p] : 0;
      if (p >= 0) s[p] = NEG_INF;
    }
    __syncthreads();
  }
}

// ---------------- rep layer 1: gather-GEMM [1024,4160]x[4160,512] via bf16 WMMA ----------------
__global__ __launch_bounds__(32) void k_rep1(const unsigned short* __restrict__ confbf,
                                             const unsigned short* __restrict__ cbf,
                                             const int* __restrict__ topidx,
                                             const unsigned short* __restrict__ w1bf,
                                             const float* __restrict__ b1,
                                             float* __restrict__ z1) {
  const int lane = threadIdx.x;
  const int ot = blockIdx.x & 31;
  const int pt = blockIdx.x >> 5;
  const int m0 = pt * 16, o0 = ot * 16;
  __shared__ int lidx[256];
  for (int i = lane; i < 256; i += 32) lidx[i] = topidx[(size_t)m0 * 16 + i];
  __syncthreads();
  const int mrow = lane & 15;
  const int koff = (lane < 16) ? 0 : 16;
  const int hi   = (lane < 16) ? 0 : 8;
  v8f acc0 = {}, acc1 = {};
  const unsigned short* wr = w1bf + (size_t)(o0 + mrow) * 4160 + koff;
  const unsigned short* cr = confbf + (size_t)(m0 + mrow) * 64;
  for (int c = 0; c < 130; c += 2) {
#pragma unroll
    for (int u = 0; u < 2; u++) {
      const int cc2 = c + u;
      const int kb = cc2 * 32 + koff;
      const unsigned short* as;
      if (kb < 64) {
        as = cr + kb;                                      // confounders region
      } else {
        const int j = kb - 64; const int rr = j >> 8; const int cc = j & 255;
        as = cbf + (size_t)lidx[mrow * 16 + rr] * 256 + cc; // gathered corpus row
      }
      v16bf av = *(const v16bf*)as;
      v16bf bv = *(const v16bf*)(wr + cc2 * 32);
      if (u == 0)
        acc0 = __builtin_amdgcn_wmma_f32_16x16x32_bf16(false, av, false, bv,
                                                       (short)0, acc0, false, false);
      else
        acc1 = __builtin_amdgcn_wmma_f32_16x16x32_bf16(false, av, false, bv,
                                                       (short)0, acc1, false, false);
    }
  }
  const float bias = b1[o0 + mrow];
#pragma unroll
  for (int r = 0; r < 8; r++)
    z1[(size_t)(m0 + r + hi) * 512 + o0 + mrow] = acc0[r] + acc1[r] + bias;
}

// ---------------- rep layer 2: [1024,512]x[512,512] via bf16 WMMA ----------------
__global__ __launch_bounds__(32) void k_rep2(const unsigned short* __restrict__ h1bf,
                                             const unsigned short* __restrict__ w2bf,
                                             const float* __restrict__ b2,
                                             float* __restrict__ z2) {
  const int lane = threadIdx.x;
  const int ot = blockIdx.x & 31;
  const int pt = blockIdx.x >> 5;
  const int m0 = pt * 16, o0 = ot * 16;
  const int mrow = lane & 15;
  const int koff = (lane < 16) ? 0 : 16;
  const int hi   = (lane < 16) ? 0 : 8;
  const unsigned short* ar = h1bf + (size_t)(m0 + mrow) * 512 + koff;
  const unsigned short* wr = w2bf + (size_t)(o0 + mrow) * 512 + koff;
  v8f acc0 = {}, acc1 = {};
#pragma unroll
  for (int c = 0; c < 8; c++) {
    v16bf av0 = *(const v16bf*)(ar + c * 32);
    v16bf bv0 = *(const v16bf*)(wr + c * 32);
    v16bf av1 = *(const v16bf*)(ar + (c + 8) * 32);
    v16bf bv1 = *(const v16bf*)(wr + (c + 8) * 32);
    acc0 = __builtin_amdgcn_wmma_f32_16x16x32_bf16(false, av0, false, bv0,
                                                   (short)0, acc0, false, false);
    acc1 = __builtin_amdgcn_wmma_f32_16x16x32_bf16(false, av1, false, bv1,
                                                   (short)0, acc1, false, false);
  }
  const float bias = b2[o0 + mrow];
#pragma unroll
  for (int r = 0; r < 8; r++)
    z2[(size_t)(m0 + r + hi) * 512 + o0 + mrow] = acc0[r] + acc1[r] + bias;
}

// ---------------- relu + layernorm over 512 ----------------
__global__ __launch_bounds__(256) void k_relu_ln(const float* __restrict__ z,
                                                 const float* __restrict__ g,
                                                 const float* __restrict__ bb,
                                                 unsigned short* __restrict__ obf,
                                                 float* __restrict__ of32, int mode) {
  const int b = blockIdx.x, t = threadIdx.x;
  const float x0 = fmaxf(z[(size_t)b * 512 + t], 0.f);
  const float x1 = fmaxf(z[(size_t)b * 512 + 256 + t], 0.f);
  __shared__ float red[256];
  red[t] = x0 + x1; __syncthreads();
  for (int off = 128; off > 0; off >>= 1) { if (t < off) red[t] += red[t + off]; __syncthreads(); }
  const float mu = red[0] * (1.0f / 512.0f);
  __syncthreads();
  const float d0 = x0 - mu, d1 = x1 - mu;
  red[t] = d0 * d0 + d1 * d1; __syncthreads();
  for (int off = 128; off > 0; off >>= 1) { if (t < off) red[t] += red[t + off]; __syncthreads(); }
  const float inv = rsqrtf(red[0] * (1.0f / 512.0f) + 1e-5f);
  const float y0 = d0 * inv * g[t] + bb[t];
  const float y1 = d1 * inv * g[256 + t] + bb[256 + t];
  if (mode == 0) {
    obf[(size_t)b * 512 + t] = f2bf(y0);
    obf[(size_t)b * 512 + 256 + t] = f2bf(y1);
  } else {
    of32[(size_t)b * 512 + t] = y0;
    of32[(size_t)b * 512 + 256 + t] = y1;
  }
}

// ---------------- tail: rep_w3 + hyp heads (factual + 2 CF) + propensity ----------------
__global__ __launch_bounds__(256) void k_tail(const float* __restrict__ h2,
                                              const float* __restrict__ treatment,
                                              const float* __restrict__ conf,
                                              const float* __restrict__ w3, const float* __restrict__ b3,
                                              const float* __restrict__ hw1, const float* __restrict__ hb1,
                                              const float* __restrict__ hw2, const float* __restrict__ hb2,
                                              const float* __restrict__ hw3, const float* __restrict__ hb3,
                                              const float* __restrict__ pw1, const float* __restrict__ pb1,
                                              const float* __restrict__ pw2, const float* __restrict__ pb2,
                                              float* __restrict__ out_factual, float* __restrict__ out_cf,
                                              float* __restrict__ out_prop, float* __restrict__ out_rep) {
  const int b = blockIdx.x, t = threadIdx.x;
  __shared__ float sh2[512], srep[256], sz[256], sz2[128], sred[256], sconf[64];
  for (int i = t; i < 512; i += 256) sh2[i] = h2[(size_t)b * 512 + i];
  for (int i = t; i < 64; i += 256) sconf[i] = conf[(size_t)b * 64 + i];
  __syncthreads();
  {
    const float* wr = w3 + (size_t)t * 512; float s = b3[t];
    for (int k = 0; k < 512; k++) s += wr[k] * sh2[k];
    srep[t] = s; out_rep[(size_t)b * 256 + t] = s;
  }
  __syncthreads();
  const float ft0 = treatment[(size_t)b * 2 + 0], ft1 = treatment[(size_t)b * 2 + 1];
  for (int cfi = 0; cfi < 3; cfi++) {
    const float t0 = (cfi == 0) ? ft0 : ((cfi == 1) ? 1.f : 0.f);
    const float t1 = (cfi == 0) ? ft1 : ((cfi == 1) ? 0.f : 1.f);
    {
      const float* wr = hw1 + (size_t)t * 258; float s = hb1[t];
      for (int k = 0; k < 256; k++) s += wr[k] * srep[k];
      s += wr[256] * t0 + wr[257] * t1;
      sz[t] = fmaxf(s, 0.f);
    }
    __syncthreads();
    if (t < 128) {
      const float* wr = hw2 + (size_t)t * 256; float s = hb2[t];
      for (int k = 0; k < 256; k++) s += wr[k] * sz[k];
      sz2[t] = fmaxf(s, 0.f);
    }
    __syncthreads();
    sred[t] = (t < 128) ? sz2[t] * hw3[t] : 0.f;
    __syncthreads();
    for (int off = 128; off > 0; off >>= 1) { if (t < off) sred[t] += sred[t + off]; __syncthreads(); }
    if (t == 0) {
      const float o = sred[0] + hb3[0];
      if (cfi == 0) out_factual[b] = o; else out_cf[(size_t)b * 2 + (cfi - 1)] = o;
    }
    __syncthreads();
  }
  {
    const float* wr = pw1 + (size_t)t * 64; float s = pb1[t];
    for (int k = 0; k < 64; k++) s += wr[k] * sconf[k];
    sz[t] = fmaxf(s, 0.f);
  }
  __syncthreads();
  if (t < 2) {
    const float* wr = pw2 + (size_t)t * 256; float s = pb2[t];
    for (int k = 0; k < 256; k++) s += wr[k] * sz[k];
    sred[t] = s;
  }
  __syncthreads();
  if (t == 0) {
    const float a = sred[0], c = sred[1];
    const float mx = fmaxf(a, c);
    const float ea = expf(a - mx), ec = expf(c - mx);
    const float inv = 1.f / (ea + ec);
    out_prop[(size_t)b * 2] = ea * inv; out_prop[(size_t)b * 2 + 1] = ec * inv;
  }
}

extern "C" void kernel_launch(void* const* d_in, const int* in_sizes, int n_in,
                              void* d_out, int out_size, void* d_ws, size_t ws_size,
                              hipStream_t stream) {
  (void)in_sizes; (void)n_in; (void)out_size; (void)ws_size;
  const float* patient   = (const float*)d_in[0];
  const float* treatment = (const float*)d_in[1];
  const float* conf      = (const float*)d_in[2];
  const float* corpus    = (const float*)d_in[3];
  const float* pe_w      = (const float*)d_in[4];
  const float* pe_b      = (const float*)d_in[5];
  const float* rep_w1    = (const float*)d_in[6];
  const float* rep_b1    = (const float*)d_in[7];
  const float* ln1_g     = (const float*)d_in[8];
  const float* ln1_b     = (const float*)d_in[9];
  const float* rep_w2    = (const float*)d_in[10];
  const float* rep_b2    = (const float*)d_in[11];
  const float* ln2_g     = (const float*)d_in[12];
  const float* ln2_b     = (const float*)d_in[13];
  const float* rep_w3    = (const float*)d_in[14];
  const float* rep_b3    = (const float*)d_in[15];
  const float* hyp_w1    = (const float*)d_in[16];
  const float* hyp_b1    = (const float*)d_in[17];
  const float* hyp_w2    = (const float*)d_in[18];
  const float* hyp_b2    = (const float*)d_in[19];
  const float* hyp_w3    = (const float*)d_in[20];
  const float* hyp_b3    = (const float*)d_in[21];
  const float* prop_w1   = (const float*)d_in[22];
  const float* prop_b1   = (const float*)d_in[23];
  const float* prop_w2   = (const float*)d_in[24];
  const float* prop_b2   = (const float*)d_in[25];

  float* out = (float*)d_out;
  float* out_factual = out;            // [1024,1]
  float* out_cf      = out + 1024;     // [1024,2,1]
  float* out_prop    = out + 3072;     // [1024,2]
  float* out_rep     = out + 5120;     // [1024,256]
  float* out_scores  = out + 267264;   // [1024,16]
  int*   out_idx     = (int*)(out + 283648); // [1024,16] int32 bits

  char* wp = (char*)d_ws;
  auto carve = [&](size_t bytes) { char* p = wp; wp += (bytes + 255) & ~(size_t)255; return p; };
  unsigned short* ws_cbf    = (unsigned short*)carve(200000ull * 256 * 2); // bf16 corpus (fits 192MB L2)
  float*          ws_invn   = (float*)carve(200000ull * 4);
  unsigned short* ws_pebf   = (unsigned short*)carve(1024ull * 256 * 2);
  unsigned short* ws_confbf = (unsigned short*)carve(1024ull * 64 * 2);
  unsigned short* ws_w1bf   = (unsigned short*)carve(512ull * 4160 * 2);
  unsigned short* ws_w2bf   = (unsigned short*)carve(512ull * 512 * 2);
  float*          ws_ps     = (float*)carve(1024ull * NSPLIT * 16 * 4);
  int*            ws_pi     = (int*)carve(1024ull * NSPLIT * 16 * 4);
  float*          ws_z1     = (float*)carve(1024ull * 512 * 4);
  unsigned short* ws_h1bf   = (unsigned short*)carve(1024ull * 512 * 2);
  float*          ws_z2     = (float*)carve(1024ull * 512 * 4);
  float*          ws_h2     = (float*)carve(1024ull * 512 * 4);

  k_corpus_prep<<<200000, 64, 0, stream>>>(corpus, ws_cbf, ws_invn);
  k_patient_emb<<<1024, 256, 0, stream>>>(patient, pe_w, pe_b, ws_pebf);
  k_f2bf<<<(512 * 4160 + 255) / 256, 256, 0, stream>>>(rep_w1, ws_w1bf, 512 * 4160);
  k_f2bf<<<(512 * 512 + 255) / 256, 256, 0, stream>>>(rep_w2, ws_w2bf, 512 * 512);
  k_f2bf<<<(1024 * 64 + 255) / 256, 256, 0, stream>>>(conf, ws_confbf, 1024 * 64);
  k_sim_topk<<<16 * NSPLIT, 128, 0, stream>>>(ws_pebf, ws_cbf, ws_invn, ws_ps, ws_pi);
  k_topk_merge<<<1024, 256, 0, stream>>>(ws_ps, ws_pi, out_scores, out_idx);
  k_rep1<<<2048, 32, 0, stream>>>(ws_confbf, ws_cbf, out_idx, ws_w1bf, rep_b1, ws_z1);
  k_relu_ln<<<1024, 256, 0, stream>>>(ws_z1, ln1_g, ln1_b, ws_h1bf, nullptr, 0);
  k_rep2<<<2048, 32, 0, stream>>>(ws_h1bf, ws_w2bf, rep_b2, ws_z2);
  k_relu_ln<<<1024, 256, 0, stream>>>(ws_z2, ln2_g, ln2_b, nullptr, ws_h2, 1);
  k_tail<<<1024, 256, 0, stream>>>(ws_h2, treatment, conf, rep_w3, rep_b3,
                                   hyp_w1, hyp_b1, hyp_w2, hyp_b2, hyp_w3, hyp_b3,
                                   prop_w1, prop_b1, prop_w2, prop_b2,
                                   out_factual, out_cf, out_prop, out_rep);
}
